// Block_88098369175928
// MI455X (gfx1250) — matrix operand order repaired and checked by reading, not compiled, not run on hardware
//
#include <hip/hip_runtime.h>
#include <hip/hip_bf16.h>
#include <math.h>

// ---------------------------------------------------------------------------
// Types
// ---------------------------------------------------------------------------
typedef __bf16 bf16;
typedef bf16  v16bf __attribute__((ext_vector_type(16)));
typedef bf16  v8bf  __attribute__((ext_vector_type(8)));
typedef float v8f   __attribute__((ext_vector_type(8)));
typedef int   v4i   __attribute__((ext_vector_type(4)));

#if __has_builtin(__builtin_amdgcn_global_load_async_to_lds_b128) && \
    __has_builtin(__builtin_amdgcn_s_wait_asynccnt)
#define USE_ASYNC_LDS 1
#else
#define USE_ASYNC_LDS 0
#endif

__device__ __forceinline__ v8f zero8() {
    v8f z;
#pragma unroll
    for (int i = 0; i < 8; ++i) z[i] = 0.0f;
    return z;
}

// CDNA5 WMMA: D(16x16 f32) = A(16x32 bf16) x B(32x16 bf16) + C
__device__ __forceinline__ v8f wmma_bf16(v16bf a, v16bf b, v8f c) {
    return __builtin_amdgcn_wmma_f32_16x16x32_bf16(
        /*neg_a=*/false, a, /*neg_b=*/false, b,
        /*c_mod=*/(short)0, c, /*reuse_a=*/false, /*reuse_b=*/false);
}

// A-fragment (16x32, bf16) from row-major [rows x ld], rows row0..+15, cols k0..+31.
// ISA 16-bit A layout: lane<16 -> row=lane, K = {k0..k0+7, k0+16..k0+23};
//                      lane>=16 -> row=lane-16, K shifted by +8.
__device__ __forceinline__ v16bf load_a_frag(const bf16* base, int ld,
                                             int row0, int k0) {
    int lane = threadIdx.x & 31;
    int r  = row0 + (lane & 15);
    int kb = k0 + ((lane >> 4) << 3);
    const bf16* p = base + (size_t)r * (size_t)ld + kb;
    v8bf lo = *(const v8bf*)(p);        // K = kb .. kb+7
    v8bf hi = *(const v8bf*)(p + 16);   // K = kb+16 .. kb+23
    v16bf f;
#pragma unroll
    for (int i = 0; i < 8; ++i) { f[i] = lo[i]; f[i + 8] = hi[i]; }
    return f;
}

// B-fragment (32x16, bf16) given B^T row-major [N x K]: rows n0..+15, cols k0..+31.
// ISA 16-bit B layout: lane<16 -> col n=lane, K = {k0..k0+7, k0+16..k0+23}; lane>=16 -> K shifted by +8.
__device__ __forceinline__ v16bf load_b_frag(const bf16* baseT, int ld,
                                             int n0, int k0) {
    int lane = threadIdx.x & 31;
    int r  = n0 + (lane & 15);
    int kb = k0 + ((lane >> 4) << 3);
    const bf16* p = baseT + (size_t)r * (size_t)ld + kb;
    v8bf lo = *(const v8bf*)(p);
    v8bf hi = *(const v8bf*)(p + 16);
    v16bf f;
#pragma unroll
    for (int i = 0; i < 8; ++i) { f[i] = lo[i]; f[i + 8] = hi[i]; }
    return f;
}

// 16-byte global -> LDS copy, async (ASYNCcnt path) when available.
// Builtin prototype (from clang diagnostic): params are v4i* in AS1 / AS3.
__device__ __forceinline__ void copy16_g2l(const bf16* g, bf16* l) {
#if USE_ASYNC_LDS
    __builtin_amdgcn_global_load_async_to_lds_b128(
        (__attribute__((address_space(1))) v4i*)g,
        (__attribute__((address_space(3))) v4i*)l, 0, 0);
#else
    *(v8bf*)l = *(const v8bf*)g;
#endif
}

__device__ __forceinline__ void wait_async_le3() {
#if USE_ASYNC_LDS
    __builtin_amdgcn_s_wait_asynccnt(3);
#endif
}
__device__ __forceinline__ void wait_async_le0() {
#if USE_ASYNC_LDS
    __builtin_amdgcn_s_wait_asynccnt(0);
#endif
}

// ---------------------------------------------------------------------------
// Elementwise / layout kernels
// ---------------------------------------------------------------------------
__global__ void cvt_f32_bf16(const float* __restrict__ in, bf16* __restrict__ out,
                             size_t n) {
    size_t i = (size_t)blockIdx.x * blockDim.x + threadIdx.x;
    if (i < n) out[i] = (bf16)in[i];
}

// W [G][K][D] f32  ->  Wt [(G*D)][K] bf16   (per-group transpose: Bt = W^T)
__global__ void transpose_w(const float* __restrict__ W, bf16* __restrict__ Wt,
                            int G, int K, int D) {
    size_t total = (size_t)G * K * D;
    size_t i = (size_t)blockIdx.x * blockDim.x + threadIdx.x;
    if (i >= total) return;
    int d  = (int)(i % D);
    size_t r = i / D;
    int kk = (int)(r % K);
    int g  = (int)(r / K);
    Wt[((size_t)(g * D + d)) * K + kk] = (bf16)W[i];
}

// V bf16 [B*T][C] (col = h*HS + d)  ->  Vt bf16 [B][H][HS][T]
__global__ void transpose_v(const bf16* __restrict__ V, bf16* __restrict__ Vt,
                            int H, int T, int C, int HS, size_t total) {
    size_t i = (size_t)blockIdx.x * blockDim.x + threadIdx.x;
    if (i >= total) return;
    int cc = (int)(i % C);
    size_t r = i / C;
    int t = (int)(r % T);
    int b = (int)(r / T);
    int h = cc / HS, d = cc % HS;
    Vt[(((size_t)(b * H + h)) * HS + d) * T + t] = V[i];
}

// ---------------------------------------------------------------------------
// WMMA GEMM with double-buffered async-LDS staging.
//   out = act(A[MxK] * Bt[NxK]^T + bias) (+ resid)
// Block = 256 threads = 8 waves as 4(M) x 2(N); wave tile 32x32;
// block tile 128(M) x 64(N); k-step 32. A/B k-slices staged in LDS
// (rows padded to 48 elems = 96B for bank spread), loads for step i+1
// issued asynchronously before the WMMAs of step i.
// ---------------------------------------------------------------------------
#define LDP 48   // padded LDS row length (elements)

__global__ __launch_bounds__(256)
void gemm_bf16_wmma(const bf16* __restrict__ A, const bf16* __restrict__ Bt,
                    const float* __restrict__ bias, const float* __restrict__ resid,
                    float* __restrict__ outF, bf16* __restrict__ outB,
                    int M, int N, int K, int relu) {
    __shared__ __align__(16) bf16 As[2][128 * LDP];
    __shared__ __align__(16) bf16 Bs[2][64 * LDP];

    int tid  = (int)threadIdx.x;
    int wave = tid >> 5;
    int lane = tid & 31;
    int wy = wave >> 1;          // 0..3 : M sub-tile
    int wx = wave & 1;           // 0..1 : N sub-tile
    int n0 = blockIdx.x * 64;
    int m0 = blockIdx.y * 128;

    // stage one 32-wide k-slice: A 128x32, B 64x32 (3 x b128 per thread)
    auto stage = [&](int buf, int k0) {
#pragma unroll
        for (int s = 0; s < 2; ++s) {
            int seg = tid + s * 256;         // 0..511
            int r = seg >> 2;                // 0..127
            int c = (seg & 3) << 3;          // 0,8,16,24
            copy16_g2l(A + (size_t)(m0 + r) * K + k0 + c,
                       &As[buf][r * LDP + c]);
        }
        {
            int r = tid >> 2;                // 0..63
            int c = (tid & 3) << 3;
            copy16_g2l(Bt + (size_t)(n0 + r) * K + k0 + c,
                       &Bs[buf][r * LDP + c]);
        }
    };

    v8f acc00 = zero8(), acc01 = zero8(), acc10 = zero8(), acc11 = zero8();

    int niter = K >> 5;
    stage(0, 0);
    for (int it = 0; it < niter; ++it) {
        int cur = it & 1;
        bool hasNext = (it + 1) < niter;
        if (hasNext) stage(1 - cur, (it + 1) << 5);   // overlap with compute below
        if (hasNext) wait_async_le3(); else wait_async_le0();
        __syncthreads();                               // cur slice visible to all

        const bf16* as = &As[cur][0];
        const bf16* bs = &Bs[cur][0];
        v16bf a0 = load_a_frag(as, LDP, wy * 32,      0);
        v16bf a1 = load_a_frag(as, LDP, wy * 32 + 16, 0);
        v16bf b0 = load_b_frag(bs, LDP, wx * 32,      0);
        v16bf b1 = load_b_frag(bs, LDP, wx * 32 + 16, 0);
        acc00 = wmma_bf16(a0, b0, acc00);
        acc01 = wmma_bf16(a0, b1, acc01);
        acc10 = wmma_bf16(a1, b0, acc10);
        acc11 = wmma_bf16(a1, b1, acc11);

        __syncthreads();                               // all reads done before overwrite
    }

    // epilogue: wave owns rows mW..mW+31, cols nW..nW+31
    int rb = ((lane >> 4) << 3);
    int cl = lane & 15;
    int mW = m0 + wy * 32;
    int nW = n0 + wx * 32;
    int c0 = nW + cl, c1 = nW + 16 + cl;
    float bi0 = bias ? bias[c0] : 0.0f;
    float bi1 = bias ? bias[c1] : 0.0f;
#pragma unroll
    for (int half = 0; half < 2; ++half) {
        v8f accL = half ? acc10 : acc00;
        v8f accR = half ? acc11 : acc01;
        int rbase = mW + half * 16 + rb;
#pragma unroll
        for (int i = 0; i < 8; ++i) {
            int r = rbase + i;
            float v0 = accL[i] + bi0;
            float v1 = accR[i] + bi1;
            if (relu) { v0 = fmaxf(v0, 0.0f); v1 = fmaxf(v1, 0.0f); }
            if (resid) {
                v0 += resid[(size_t)r * N + c0];
                v1 += resid[(size_t)r * N + c1];
            }
            if (outF) {
                outF[(size_t)r * N + c0] = v0;
                outF[(size_t)r * N + c1] = v1;
            }
            if (outB) {
                outB[(size_t)r * N + c0] = (bf16)v0;
                outB[(size_t)r * N + c1] = (bf16)v1;
            }
        }
    }
}

// ---------------------------------------------------------------------------
// Flash-style causal attention. One wave per (b, h, 16-row Q tile).
// Q,K bf16 laid out [B][T][C] at column h*HS; Vt bf16 [B][H][HS][T].
// Online softmax; P goes C-layout -> LDS -> A-layout for the P*V WMMA.
// ---------------------------------------------------------------------------
__global__ __launch_bounds__(32)
void attn_fwd(const bf16* __restrict__ Q, const bf16* __restrict__ Kb,
              const bf16* __restrict__ Vt, bf16* __restrict__ Y,
              int H, int T, int C, int HS, float scale) {
    __shared__ __align__(16) bf16 plds[16 * 32];

    int lane = (int)(threadIdx.x & 31);
    int bh = blockIdx.y;
    int b = bh / H, h = bh % H;
    int m0 = blockIdx.x * 16;

    const bf16* q  = Q  + (size_t)b * T * C + (size_t)h * HS;
    const bf16* k  = Kb + (size_t)b * T * C + (size_t)h * HS;
    const bf16* vt = Vt + (size_t)bh * HS * T;
    bf16*       y  = Y  + (size_t)b * T * C + (size_t)h * HS;

    v16bf qa0 = load_a_frag(q, C, m0, 0);
    v16bf qa1 = load_a_frag(q, C, m0, 32);

    v8f o[4];
#pragma unroll
    for (int dt = 0; dt < 4; ++dt) o[dt] = zero8();
    float mrow[8], lrow[8];
#pragma unroll
    for (int i = 0; i < 8; ++i) { mrow[i] = -INFINITY; lrow[i] = 0.0f; }

    int rb  = ((lane >> 4) << 3);
    int col = (lane & 15);
    int nst = (m0 + 15) / 32 + 1;   // causal: s-tiles of 32 covering s <= m0+15

    for (int st = 0; st < nst; ++st) {
        int s0 = st * 32;
        v16bf kb00 = load_b_frag(k, C, s0,      0);
        v16bf kb01 = load_b_frag(k, C, s0,      32);
        v16bf kb10 = load_b_frag(k, C, s0 + 16, 0);
        v16bf kb11 = load_b_frag(k, C, s0 + 16, 32);
        v8f sc0 = zero8(), sc1 = zero8();
        sc0 = wmma_bf16(qa0, kb00, sc0);
        sc0 = wmma_bf16(qa1, kb01, sc0);
        sc1 = wmma_bf16(qa0, kb10, sc1);
        sc1 = wmma_bf16(qa1, kb11, sc1);

#pragma unroll
        for (int i = 0; i < 8; ++i) {
            int t = m0 + rb + i;
            float a0 = sc0[i] * scale;
            float a1 = sc1[i] * scale;
            if (s0 + col      > t) a0 = -INFINITY;
            if (s0 + 16 + col > t) a1 = -INFINITY;
            sc0[i] = a0; sc1[i] = a1;
        }

#pragma unroll
        for (int i = 0; i < 8; ++i) {
            float tmax = fmaxf(sc0[i], sc1[i]);
#pragma unroll
            for (int off = 8; off >= 1; off >>= 1)
                tmax = fmaxf(tmax, __shfl_xor(tmax, off, 32));
            float mnew = fmaxf(mrow[i], tmax);
            float cf = __expf(mrow[i] - mnew);
            float p0 = __expf(sc0[i] - mnew);
            float p1 = __expf(sc1[i] - mnew);
            float rsum = p0 + p1;
#pragma unroll
            for (int off = 8; off >= 1; off >>= 1)
                rsum += __shfl_xor(rsum, off, 32);
            lrow[i] = lrow[i] * cf + rsum;
            mrow[i] = mnew;
            sc0[i] = p0; sc1[i] = p1;
#pragma unroll
            for (int dt = 0; dt < 4; ++dt) o[dt][i] *= cf;
        }

        // P: C-layout -> LDS (row-major 16x32) -> A-layout fragment
#pragma unroll
        for (int i = 0; i < 8; ++i) {
            plds[(rb + i) * 32 + col]      = (bf16)sc0[i];
            plds[(rb + i) * 32 + col + 16] = (bf16)sc1[i];
        }
        __syncthreads();
        v16bf pa = load_a_frag(plds, 32, 0, 0);
        __syncthreads();

#pragma unroll
        for (int dt = 0; dt < 4; ++dt) {
            v16bf vb = load_b_frag(vt, T, dt * 16, s0);
            o[dt] = wmma_bf16(pa, vb, o[dt]);
        }
    }

#pragma unroll
    for (int i = 0; i < 8; ++i) {
        float inv = 1.0f / lrow[i];
        int r = m0 + rb + i;
#pragma unroll
        for (int dt = 0; dt < 4; ++dt) {
            y[(size_t)r * C + dt * 16 + col] = (bf16)(o[dt][i] * inv);
        }
    }
}

// ---------------------------------------------------------------------------
// Host launch
// ---------------------------------------------------------------------------
extern "C" void kernel_launch(void* const* d_in, const int* in_sizes, int n_in,
                              void* d_out, int out_size, void* d_ws, size_t ws_size,
                              hipStream_t stream) {
    (void)in_sizes; (void)n_in; (void)out_size; (void)ws_size;
    const int B = 4, T = 2048, C = 1024, H = 16, HS = 64, FF = 4096;
    const int M = B * T;               // 8192 rows
    const float scale = 0.03125f;      // C^-0.5 (reference uses full embed dim)

    const float* x     = (const float*)d_in[0];
    const float* Wq    = (const float*)d_in[1];
    const float* Wk    = (const float*)d_in[2];
    const float* Wv    = (const float*)d_in[3];
    const float* Wproj = (const float*)d_in[4];
    const float* bproj = (const float*)d_in[5];
    const float* W1    = (const float*)d_in[6];
    const float* b1    = (const float*)d_in[7];
    const float* W2    = (const float*)d_in[8];
    const float* b2    = (const float*)d_in[9];

    char* ws = (char*)d_ws;
    size_t off = 0;
    auto alloc = [&](size_t bytes) -> char* {
        char* p = ws + off;
        off += (bytes + 255) & ~(size_t)255;
        return p;
    };
    const size_t MTC = (size_t)M * C;
    bf16* xbf    = (bf16*)alloc(MTC * 2);
    bf16* Wqt    = (bf16*)alloc((size_t)C * C * 2);
    bf16* Wkt    = (bf16*)alloc((size_t)C * C * 2);
    bf16* Wvt    = (bf16*)alloc((size_t)C * C * 2);
    bf16* Wpt    = (bf16*)alloc((size_t)C * C * 2);
    bf16* W1t    = (bf16*)alloc((size_t)C * FF * 2);
    bf16* W2t    = (bf16*)alloc((size_t)FF * C * 2);
    bf16* qbf    = (bf16*)alloc(MTC * 2);
    bf16* kbf    = (bf16*)alloc(MTC * 2);
    bf16* vbf    = (bf16*)alloc(MTC * 2);
    bf16* vT     = (bf16*)alloc(MTC * 2);
    bf16* ybf    = (bf16*)alloc(MTC * 2);
    float* x2f   = (float*)alloc(MTC * 4);
    bf16* x2bf   = (bf16*)alloc(MTC * 2);
    bf16* h1bf   = (bf16*)alloc((size_t)M * FF * 2);

    {
        size_t n = MTC;
        cvt_f32_bf16<<<(unsigned)((n + 255) / 256), 256, 0, stream>>>(x, xbf, n);
        size_t nw = (size_t)C * C;
        transpose_w<<<(unsigned)((nw + 255) / 256), 256, 0, stream>>>(Wq, Wqt, H, C, HS);
        transpose_w<<<(unsigned)((nw + 255) / 256), 256, 0, stream>>>(Wk, Wkt, H, C, HS);
        transpose_w<<<(unsigned)((nw + 255) / 256), 256, 0, stream>>>(Wv, Wvt, H, C, HS);
        transpose_w<<<(unsigned)((nw + 255) / 256), 256, 0, stream>>>(Wproj, Wpt, 1, C, C);
        size_t n1 = (size_t)C * FF;
        transpose_w<<<(unsigned)((n1 + 255) / 256), 256, 0, stream>>>(W1, W1t, 1, C, FF);
        transpose_w<<<(unsigned)((n1 + 255) / 256), 256, 0, stream>>>(W2, W2t, 1, FF, C);
    }

    dim3 gC(C / 64, M / 128);    // N=1024 GEMMs: 16 x 64 blocks
    dim3 gF(FF / 64, M / 128);   // N=4096 GEMM:  64 x 64 blocks

    // QKV projections -> bf16 [B*T][C] (col = h*HS+d)
    gemm_bf16_wmma<<<gC, 256, 0, stream>>>(xbf, Wqt, nullptr, nullptr, nullptr, qbf, M, C, C, 0);
    gemm_bf16_wmma<<<gC, 256, 0, stream>>>(xbf, Wkt, nullptr, nullptr, nullptr, kbf, M, C, C, 0);
    gemm_bf16_wmma<<<gC, 256, 0, stream>>>(xbf, Wvt, nullptr, nullptr, nullptr, vbf, M, C, C, 0);

    // V -> Vt [B][H][HS][T]
    transpose_v<<<(unsigned)((MTC + 255) / 256), 256, 0, stream>>>(vbf, vT, H, T, C, HS, MTC);

    // causal flash attention -> ybf
    dim3 ga(T / 16, B * H);
    attn_fwd<<<ga, 32, 0, stream>>>(qbf, kbf, vT, ybf, H, T, C, HS, scale);

    // output projection + bias + residual(x) -> x2 (f32 + bf16)
    gemm_bf16_wmma<<<gC, 256, 0, stream>>>(ybf, Wpt, bproj, x, x2f, x2bf, M, C, C, 0);

    // FFN1: relu(x2 @ W1 + b1) -> h1 bf16
    gemm_bf16_wmma<<<gF, 256, 0, stream>>>(x2bf, W1t, b1, nullptr, nullptr, h1bf, M, FF, C, 1);

    // FFN2: h1 @ W2 + b2 + x2 -> d_out (f32)
    gemm_bf16_wmma<<<gC, 256, 0, stream>>>(h1bf, W2t, b2, x2f, (float*)d_out, nullptr, M, C, FF, 0);
}
